// RNN_30837865185540
// MI455X (gfx1250) — matrix operand order repaired and checked by reading, not compiled, not run on hardware
//
#include <hip/hip_runtime.h>
#include <hip/hip_bf16.h>

// ---------------------------------------------------------------------------
// Types for CDNA5 WMMA (wave32)
// ---------------------------------------------------------------------------
typedef __attribute__((ext_vector_type(16))) __bf16 v16bf;
typedef __attribute__((ext_vector_type(8)))  __bf16 v8bf;
typedef __attribute__((ext_vector_type(8)))  float  v8f;

#define NB 4096   // batch
#define NT 64     // timesteps
#define NF 80     // features
#define NH 80     // hidden
#define AK 192    // padded K for LSTM A/B: [x:0..79][pad][h:96..175][pad]

__device__ __forceinline__ unsigned short f2bf(float f) {
    union { float f; unsigned u; } x; x.f = f;
    unsigned r = x.u + 0x7FFFu + ((x.u >> 16) & 1u);   // round-to-nearest-even
    return (unsigned short)(r >> 16);
}
__device__ __forceinline__ float fast_rcp(float x) {
    return __builtin_amdgcn_rcpf(x);                   // v_rcp_f32
}
__device__ __forceinline__ float sigmoidf(float x) {
    return fast_rcp(1.0f + __expf(-x));
}
__device__ __forceinline__ float tanh_fast(float x) {
#if __has_builtin(__builtin_amdgcn_tanhf)
    return __builtin_amdgcn_tanhf(x);                  // v_tanh_f32 (CDNA5 trans op)
#else
    float e = __expf(-2.0f * x);
    return (1.0f - e) * fast_rcp(1.0f + e);
#endif
}

// A fragment (16x32 bf16, MxK): lane 0-15 -> M=lane, K = kk*32 + {0..7,16..23}
//                               lane 16-31 -> M=lane-16, K = kk*32 + {8..15,24..31}
__device__ __forceinline__ v16bf ld_a(const unsigned short* base, int stride,
                                      int kk, int lane) {
    int hi = (lane >> 4) & 1;
    int m  = lane & 15;
    const unsigned short* p = base + m * stride + kk * 32 + hi * 8;
    v8bf lo = *(const v8bf*)p;
    v8bf hv = *(const v8bf*)(p + 16);
    return __builtin_shufflevector(lo, hv, 0,1,2,3,4,5,6,7,8,9,10,11,12,13,14,15);
}

// B fragment (32x16 bf16, KxN) stored as Bmat[n][k] (row = output col, K-major):
// lane 0-15 -> N=n0+lane, K = kk*32+{0..7,16..23} ; lane 16-31 -> N=n0+lane-16, K = kk*32+{8..15,24..31}
__device__ __forceinline__ v16bf ld_b(const unsigned short* base, int stride,
                                      int n0, int kk, int lane) {
    int hi = (lane >> 4) & 1;
    const unsigned short* p = base + (n0 + (lane & 15)) * stride + kk * 32 + hi * 8;
    v8bf lo = *(const v8bf*)p;
    v8bf hv = *(const v8bf*)(p + 16);
    return __builtin_shufflevector(lo, hv, 0,1,2,3,4,5,6,7,8,9,10,11,12,13,14,15);
}

__device__ __forceinline__ v8f wmma_bf16(v16bf a, v16bf b, v8f c) {
    return __builtin_amdgcn_wmma_f32_16x16x32_bf16(false, a, false, b,
                                                   (short)0, c, false, false);
}

// ---------------------------------------------------------------------------
// Prep: convert/pad all weights to bf16 B-operand layout in workspace
//   W1b [240][256], W2b [80][960], Bw [320][192], biasSum [320]
// ---------------------------------------------------------------------------
__global__ void prep_kernel(const float* __restrict__ c1w,
                            const float* __restrict__ c2w,
                            const float* __restrict__ w_ih,
                            const float* __restrict__ w_hh,
                            const float* __restrict__ b_ih,
                            const float* __restrict__ b_hh,
                            unsigned short* __restrict__ W1b,
                            unsigned short* __restrict__ W2b,
                            unsigned short* __restrict__ Bw,
                            float* __restrict__ biasSum) {
    int i = blockIdx.x * blockDim.x + threadIdx.x;
    if (i < 240 * 256) {
        int n = i >> 8, k = i & 255;
        float v = (k < 240) ? c1w[n * 240 + k] : 0.0f;
        W1b[i] = f2bf(v);
        return;
    }
    i -= 240 * 256;
    if (i < 80 * 960) { W2b[i] = f2bf(c2w[i]); return; }
    i -= 80 * 960;
    if (i < 320 * AK) {
        int n = i / AK, k = i % AK;
        float v = 0.0f;
        if (k < 80)                  v = w_ih[n * 80 + k];
        else if (k >= 96 && k < 176) v = w_hh[n * 80 + (k - 96)];
        Bw[i] = f2bf(v);
        return;
    }
    i -= 320 * AK;
    if (i < 320) biasSum[i] = b_ih[i] + b_hh[i];
}

// ---------------------------------------------------------------------------
// Stage 1: 1x1 conv 240->240 + sigmoid, as GEMM [16384,240]x[240,240]
// rows = (b, pos) ; writes X1[b][c2*4+pos] bf16 (A-layout for stage 2)
// ---------------------------------------------------------------------------
__global__ __launch_bounds__(160) void conv1_kernel(
    const float* __restrict__ timeless,        // [B,240,2,2]
    const unsigned short* __restrict__ W1b,    // [240][256]
    const float* __restrict__ conv1_b,         // [240]
    unsigned short* __restrict__ X1) {         // [B][960] bf16
    __shared__ unsigned short Ash[16 * 256];
    const int tid = threadIdx.x, lane = tid & 31, wv = tid >> 5;
    const int R0 = blockIdx.x * 16;
    for (int i = tid; i < 16 * 256; i += 160) {
        int r = i >> 8, c = i & 255;
        int R = R0 + r;
        float v = (c < 240) ? timeless[(size_t)(R >> 2) * 960 + c * 4 + (R & 3)] : 0.0f;
        Ash[r * 256 + c] = f2bf(v);
    }
    __syncthreads();
    const int hi = (lane >> 4) & 1, col = lane & 15;
#pragma unroll
    for (int j = 0; j < 3; ++j) {
        int n0 = (wv * 3 + j) * 16;
        v8f acc = {0.f,0.f,0.f,0.f,0.f,0.f,0.f,0.f};
#pragma unroll
        for (int kk = 0; kk < 8; ++kk) {
            v16bf a = ld_a(Ash, 256, kk, lane);
            v16bf b = ld_b(W1b, 256, n0, kk, lane);
            acc = wmma_bf16(a, b, acc);
        }
        float bb = conv1_b[n0 + col];
#pragma unroll
        for (int v = 0; v < 8; ++v) {
            int R = R0 + v + 8 * hi;
            float val = sigmoidf(acc[v] + bb);
            X1[(size_t)(R >> 2) * 960 + (n0 + col) * 4 + (R & 3)] = f2bf(val);
        }
    }
}

// ---------------------------------------------------------------------------
// Stage 2: 2x2 full-contraction conv + sigmoid, as GEMM [4096,960]x[960,80]
// ---------------------------------------------------------------------------
__global__ __launch_bounds__(160) void conv2_kernel(
    const unsigned short* __restrict__ X1,     // [B][960] bf16
    const unsigned short* __restrict__ W2b,    // [80][960]
    const float* __restrict__ conv2_b,         // [80]
    float* __restrict__ rosters) {             // [B][80]
    __shared__ unsigned short Ash[16 * 960];
    const int tid = threadIdx.x, lane = tid & 31, wv = tid >> 5;
    const int b0 = blockIdx.x * 16;
    for (int i = tid; i < 16 * 960 / 8; i += 160) {          // 8-elem vector copies
        int r = i / 120, cc = (i % 120) * 8;
        *(v8bf*)(Ash + r * 960 + cc) =
            *(const v8bf*)(X1 + (size_t)(b0 + r) * 960 + cc);
    }
    __syncthreads();
    const int hi = (lane >> 4) & 1, col = lane & 15;
    const int n0 = wv * 16;
    v8f acc = {0.f,0.f,0.f,0.f,0.f,0.f,0.f,0.f};
#pragma unroll
    for (int kk = 0; kk < 30; ++kk) {
        v16bf a = ld_a(Ash, 960, kk, lane);
        v16bf b = ld_b(W2b, 960, n0, kk, lane);
        acc = wmma_bf16(a, b, acc);
    }
    float bb = conv2_b[n0 + col];
#pragma unroll
    for (int v = 0; v < 8; ++v)
        rosters[(size_t)(b0 + v + 8 * hi) * NH + n0 + col] = sigmoidf(acc[v] + bb);
}

// ---------------------------------------------------------------------------
// LSTM: each block owns 16 batch rows across all T steps (no global sync).
// 5 waves; wave w owns hidden columns [w*16, w*16+16) -> 4 gate tiles.
// ---------------------------------------------------------------------------
__device__ __forceinline__ void fc_partials(const float h[8], float fw0, float fw1,
                                            float* part, int wv, int lane, int hi) {
#pragma unroll
    for (int v = 0; v < 8; ++v) {
        float a0 = h[v] * fw0, a1 = h[v] * fw1;
#pragma unroll
        for (int m = 1; m < 16; m <<= 1) {       // reduce across the 16 columns
            a0 += __shfl_xor(a0, m, 32);
            a1 += __shfl_xor(a1, m, 32);
        }
        if ((lane & 15) == 0) {
            int row = v + 8 * hi;
            part[(wv * 16 + row) * 2 + 0] = a0;
            part[(wv * 16 + row) * 2 + 1] = a1;
        }
    }
}

__device__ __forceinline__ void fill_x(unsigned short* A, const float* time,
                                       int b0, int t, int tid) {
    int e = tid * 8;                 // 160 threads * 8 = 1280 = 16*80 exactly
    int r = e / NF, c = e % NF;
    const float* src = time + (size_t)(b0 + r) * NT * NF + (size_t)t * NF + c;
    float4 s0 = *(const float4*)(src);
    float4 s1 = *(const float4*)(src + 4);
    unsigned short* dst = A + r * AK + c;
    dst[0] = f2bf(s0.x); dst[1] = f2bf(s0.y); dst[2] = f2bf(s0.z); dst[3] = f2bf(s0.w);
    dst[4] = f2bf(s1.x); dst[5] = f2bf(s1.y); dst[6] = f2bf(s1.z); dst[7] = f2bf(s1.w);
}

__global__ __launch_bounds__(160) void lstm_kernel(
    const float* __restrict__ time,            // [B,T,F]
    const float* __restrict__ rosters,         // [B,H]
    const unsigned short* __restrict__ Bw,     // [320][192] bf16
    const float* __restrict__ biasSum,         // [320]
    const float* __restrict__ fc_w,            // [2,80]
    const float* __restrict__ fc_b,            // [2]
    float* __restrict__ out) {                 // [65,B,2]
    extern __shared__ char smem[];
    unsigned short* A = (unsigned short*)smem;                      // 16*192
    unsigned short* W = (unsigned short*)(smem + 16 * AK * 2);      // 320*192
    float* part = (float*)(smem + 16 * AK * 2 + 320 * AK * 2);      // [5*16][2]

    const int tid = threadIdx.x, lane = tid & 31, wv = tid >> 5;
    const int b0 = blockIdx.x * 16;
    const int hi = (lane >> 4) & 1, col = lane & 15;

    // per-thread source pointer for x staging (used for prefetch too)
    const int xr = (tid * 8) / NF, xc = (tid * 8) % NF;
    const float* xsrc = time + (size_t)(b0 + xr) * NT * NF + xc;

    // stage weights into LDS once (reused by 64 steps)
    {
        const v8bf* src = (const v8bf*)Bw;
        v8bf* dst = (v8bf*)W;
        for (int i = tid; i < 320 * AK / 8; i += 160) dst[i] = src[i];
    }
    // zero K-padding columns of A (80..95 and 176..191)
    for (int i = tid; i < 16 * 32; i += 160) {
        int r = i >> 5, c = i & 31;
        A[r * AK + ((c < 16) ? (80 + c) : (160 + c))] = 0;
    }

    const float fw0 = fc_w[wv * 16 + col];
    const float fw1 = fc_w[NH + wv * 16 + col];
    const float fb0 = fc_b[0], fb1 = fc_b[1];
    const float bI = biasSum[0 * NH + wv * 16 + col];
    const float bF = biasSum[1 * NH + wv * 16 + col];
    const float bG = biasSum[2 * NH + wv * 16 + col];
    const float bO = biasSum[3 * NH + wv * 16 + col];

    // init h = c = rosters; stage h into A, FC partials for out[0]
    float creg[8], hreg[8];
#pragma unroll
    for (int v = 0; v < 8; ++v) {
        int row = v + 8 * hi;
        float rv = rosters[(size_t)(b0 + row) * NH + wv * 16 + col];
        creg[v] = rv; hreg[v] = rv;
        A[row * AK + 96 + wv * 16 + col] = f2bf(rv);
    }
    fc_partials(hreg, fw0, fw1, part, wv, lane, hi);
    fill_x(A, time, b0, 0, tid);
    __syncthreads();

    for (int t = 0; t < NT; ++t) {
        if (wv == 0) {   // emit out[t] from h_t partials (h_0 = rosters)
            int row = lane >> 1, k = lane & 1;
            float s = k ? fb1 : fb0;
#pragma unroll
            for (int w2 = 0; w2 < 5; ++w2) s += part[(w2 * 16 + row) * 2 + k];
            out[(size_t)t * NB * 2 + (size_t)(b0 + row) * 2 + k] = s;
        }
        // prefetch next timestep's x row into cache (overlaps the WMMAs)
        if (t + 1 < NT)
            __builtin_prefetch(xsrc + (size_t)(t + 1) * NF, 0, 3);

        // gates = [x_t | h_t] @ [w_ih ; w_hh]^T   (K = 192, 6 chunks of 32)
        v8f acc0 = {0.f,0.f,0.f,0.f,0.f,0.f,0.f,0.f};
        v8f acc1 = acc0, acc2 = acc0, acc3 = acc0;
#pragma unroll
        for (int kk = 0; kk < 6; ++kk) {
            v16bf a  = ld_a(A, AK, kk, lane);
            v16bf b0f = ld_b(W, AK, 0 * NH + wv * 16, kk, lane);
            acc0 = wmma_bf16(a, b0f, acc0);
            v16bf b1f = ld_b(W, AK, 1 * NH + wv * 16, kk, lane);
            acc1 = wmma_bf16(a, b1f, acc1);
            v16bf b2f = ld_b(W, AK, 2 * NH + wv * 16, kk, lane);
            acc2 = wmma_bf16(a, b2f, acc2);
            v16bf b3f = ld_b(W, AK, 3 * NH + wv * 16, kk, lane);
            acc3 = wmma_bf16(a, b3f, acc3);
        }
#pragma unroll
        for (int v = 0; v < 8; ++v) {
            float iv = sigmoidf(acc0[v] + bI);
            float fv = sigmoidf(acc1[v] + bF);
            float gv = tanh_fast(acc2[v] + bG);
            float ov = sigmoidf(acc3[v] + bO);
            float cv = fv * creg[v] + iv * gv;
            creg[v] = cv;
            hreg[v] = ov * tanh_fast(cv);
        }
        __syncthreads();                       // all WMMA reads of A done
#pragma unroll
        for (int v = 0; v < 8; ++v)            // h_{t+1} -> A (bf16)
            A[(v + 8 * hi) * AK + 96 + wv * 16 + col] = f2bf(hreg[v]);
        fc_partials(hreg, fw0, fw1, part, wv, lane, hi);
        if (t + 1 < NT) fill_x(A, time, b0, t + 1, tid);
        __syncthreads();                       // A/partials ready for next step
    }
    if (wv == 0) {                             // out[64] from h_64
        int row = lane >> 1, k = lane & 1;
        float s = k ? fb1 : fb0;
#pragma unroll
        for (int w2 = 0; w2 < 5; ++w2) s += part[(w2 * 16 + row) * 2 + k];
        out[(size_t)NT * NB * 2 + (size_t)(b0 + row) * 2 + k] = s;
    }
}

// ---------------------------------------------------------------------------
// Host launcher
// ---------------------------------------------------------------------------
extern "C" void kernel_launch(void* const* d_in, const int* in_sizes, int n_in,
                              void* d_out, int out_size, void* d_ws, size_t ws_size,
                              hipStream_t stream) {
    (void)in_sizes; (void)n_in; (void)out_size; (void)ws_size;
    const float* timeless = (const float*)d_in[0];
    const float* time     = (const float*)d_in[1];
    const float* conv1_w  = (const float*)d_in[2];
    const float* conv1_b  = (const float*)d_in[3];
    const float* conv2_w  = (const float*)d_in[4];
    const float* conv2_b  = (const float*)d_in[5];
    const float* w_ih     = (const float*)d_in[6];
    const float* w_hh     = (const float*)d_in[7];
    const float* b_ih     = (const float*)d_in[8];
    const float* b_hh     = (const float*)d_in[9];
    const float* fc_w     = (const float*)d_in[10];
    const float* fc_b     = (const float*)d_in[11];
    float* out = (float*)d_out;

    char* ws = (char*)d_ws;
    const size_t OFF_ROSTERS = 0;                              // 4096*80*4
    const size_t OFF_X1   = OFF_ROSTERS + (size_t)NB * NH * 4; // 4096*960*2
    const size_t OFF_W1B  = OFF_X1  + (size_t)NB * 960 * 2;    // 240*256*2
    const size_t OFF_W2B  = OFF_W1B + (size_t)240 * 256 * 2;   // 80*960*2
    const size_t OFF_BW   = OFF_W2B + (size_t)80 * 960 * 2;    // 320*192*2
    const size_t OFF_BIAS = OFF_BW  + (size_t)320 * AK * 2;    // 320*4

    float*          rosters = (float*)(ws + OFF_ROSTERS);
    unsigned short* X1      = (unsigned short*)(ws + OFF_X1);
    unsigned short* W1b     = (unsigned short*)(ws + OFF_W1B);
    unsigned short* W2b     = (unsigned short*)(ws + OFF_W2B);
    unsigned short* Bw      = (unsigned short*)(ws + OFF_BW);
    float*          biasSum = (float*)(ws + OFF_BIAS);

    const int prep_total = 240 * 256 + 80 * 960 + 320 * AK + 320;
    prep_kernel<<<(prep_total + 255) / 256, 256, 0, stream>>>(
        conv1_w, conv2_w, w_ih, w_hh, b_ih, b_hh, W1b, W2b, Bw, biasSum);

    conv1_kernel<<<(NB * 4) / 16, 160, 0, stream>>>(timeless, W1b, conv1_b, X1);
    conv2_kernel<<<NB / 16, 160, 0, stream>>>(X1, W2b, conv2_b, rosters);

    const int lstm_smem = 16 * AK * 2 + 320 * AK * 2 + 5 * 16 * 2 * 4;  // 129664 B
    (void)hipFuncSetAttribute((const void*)lstm_kernel,
                              hipFuncAttributeMaxDynamicSharedMemorySize, lstm_smem);
    lstm_kernel<<<NB / 16, 160, lstm_smem, stream>>>(
        time, rosters, Bw, biasSum, fc_w, fc_b, out);
}
